// NoiseBandNet_6665789243751
// MI455X (gfx1250) — matrix operand order repaired and checked by reading, not compiled, not run on hardware
//
#include <hip/hip_runtime.h>

// ---------------------------------------------------------------------------
// NoiseBandNet forward for gfx1250 (MI455X, wave32, WMMA).
// ---------------------------------------------------------------------------

typedef __bf16 bf16;
typedef __attribute__((ext_vector_type(16))) __bf16 v16bf;
typedef __attribute__((ext_vector_type(8)))  __bf16 bf16x8;
typedef __attribute__((ext_vector_type(8)))  float  v8f;

#define HID        128
#define NBAND      512
#define NOISE_LEN  65536
#define SYNTH_WIN  32
#define FRAME_LEN  2048
#define BATCH      2
#define TLEN       2048
#define MTOT       (BATCH * TLEN)   // 4096 rows through all GEMMs

// ---------------------------------------------------------------------------
// Weight convert with optional zero-pad of the K dimension (keeps K % 32 == 0)
// ---------------------------------------------------------------------------
__global__ void pad_convert_bf16_kernel(const float* __restrict__ src, bf16* __restrict__ dst,
                                        int rows, int src_cols, int dst_cols) {
    int i = blockIdx.x * blockDim.x + threadIdx.x;
    int total = rows * dst_cols;
    if (i >= total) return;
    int r = i / dst_cols, c = i % dst_cols;
    float v = (c < src_cols) ? src[r * src_cols + c] : 0.0f;
    dst[i] = (bf16)v;
}

__global__ void zero_bf16_kernel(bf16* __restrict__ dst, int n) {
    int i = blockIdx.x * blockDim.x + threadIdx.x;
    if (i < n) dst[i] = (bf16)0.0f;
}

// ---------------------------------------------------------------------------
// Input MLPs: Linear(1->128) + LayerNorm + LeakyReLU(0.01), both control
// params, concatenated into hidden bf16 [MTOT, 256].
// ---------------------------------------------------------------------------
__device__ inline void ln_stats128(float v, float* red, float& mean, float& rstd) {
    int j = threadIdx.x;
    red[j] = v; __syncthreads();
    for (int o = 64; o > 0; o >>= 1) { if (j < o) red[j] += red[j + o]; __syncthreads(); }
    mean = red[0] * (1.0f / 128.0f); __syncthreads();
    float d = v - mean;
    red[j] = d * d; __syncthreads();
    for (int o = 64; o > 0; o >>= 1) { if (j < o) red[j] += red[j + o]; __syncthreads(); }
    rstd = __frsqrt_rn(red[0] * (1.0f / 128.0f) + 1e-5f); __syncthreads();
}

__global__ void input_mlp_kernel(const float* __restrict__ cp0, const float* __restrict__ cp1,
                                 const float* __restrict__ w0, const float* __restrict__ b0,
                                 const float* __restrict__ g0, const float* __restrict__ be0,
                                 const float* __restrict__ w1, const float* __restrict__ b1,
                                 const float* __restrict__ g1, const float* __restrict__ be1,
                                 bf16* __restrict__ hidden) {
    __shared__ float red[128];
    int bt = blockIdx.x;       // 0..MTOT-1, row = b*TLEN + t
    int j = threadIdx.x;       // 0..127
    float x0 = cp0[bt], x1 = cp1[bt];
    float y0 = fmaf(w0[j], x0, b0[j]);
    float y1 = fmaf(w1[j], x1, b1[j]);
    float m, r;
    ln_stats128(y0, red, m, r);
    float z0 = (y0 - m) * r * g0[j] + be0[j];
    z0 = (z0 >= 0.0f) ? z0 : 0.01f * z0;
    ln_stats128(y1, red, m, r);
    float z1 = (y1 - m) * r * g1[j] + be1[j];
    z1 = (z1 >= 0.0f) ? z1 : 0.01f * z1;
    hidden[bt * 256 + j]       = (bf16)z0;
    hidden[bt * 256 + 128 + j] = (bf16)z1;
}

// ---------------------------------------------------------------------------
// Generic WMMA bf16 GEMM:  out[16 x N] = A[16 x K] @ W^T + bias  per block.
//   A: bf16 activations [MTOT, lda], row-major.  W: bf16 [N, K] row-major.
//   mode 0: f32 out (ld = N)                         -> GRU input projection
//   mode 1: LayerNorm + LeakyReLU, bf16 out (out_ld) -> output MLP layers
//   mode 2: 2*sigmoid(x)^ln10 + eps, f32 transposed store amps[b][n][t]
// Block = 256 threads = 8 waves; wave w owns n-tiles w, w+8, ...
// Fragment layouts per CDNA5 ISA 7.12.2 (16-bit A 16x32, B 32x16, f32 C/D).
// ---------------------------------------------------------------------------
__global__ void wmma_gemm_kernel(const bf16* __restrict__ Abuf, int lda,
                                 const bf16* __restrict__ Wbuf,
                                 const float* __restrict__ bias,
                                 const float* __restrict__ gamma,
                                 const float* __restrict__ beta,
                                 void* __restrict__ outp,
                                 int K, int N, int out_ld, int mode) {
    extern __shared__ char smem[];
    bf16*  As = (bf16*)smem;                       // [16][K]
    float* Os = (float*)(smem + 16 * K * 2);       // [16][N]  (mode 1 only)
    __shared__ float mrow[16], rrow[16];

    const int m0   = blockIdx.x * 16;
    const int tid  = threadIdx.x;
    const int wave = tid >> 5;
    const int lane = tid & 31;
    const int hs   = lane >> 4;     // lane half-select
    const int lr   = lane & 15;

    // Stage the 16 x K activation block in LDS (coalesced).
    for (int i = tid; i < 16 * K; i += 256) {
        int r = i / K, c = i % K;
        As[i] = Abuf[(m0 + r) * lda + c];
    }
    __syncthreads();

    const int nk = K >> 5;          // K chunks of 32
    const int ntiles = N >> 4;

    for (int nt = wave; nt < ntiles; nt += 8) {
        v8f acc = {};
        const int n = nt * 16 + lr;
        const bf16* Wrow = Wbuf + n * K + hs * 8;    // B frag: K = kc*32 + hs*8 + {0..7, 16..23}
        const bf16* Arow = As + lr * K + hs * 8;     // A frag: K = kc*32 + hs*8 + {0..7, 16..23}
        for (int kc = 0; kc < nk; ++kc) {
            union { bf16x8 h[2]; v16bf v; } af, bfr;
            af.h[0]  = *(const bf16x8*)(Arow + kc * 32);
            af.h[1]  = *(const bf16x8*)(Arow + kc * 32 + 16);
            bfr.h[0] = *(const bf16x8*)(Wrow + kc * 32);
            bfr.h[1] = *(const bf16x8*)(Wrow + kc * 32 + 16);
            acc = __builtin_amdgcn_wmma_f32_16x16x32_bf16(
                false, af.v, false, bfr.v, (short)0, acc, false, false);
        }
        const float bn = bias ? bias[n] : 0.0f;
        if (mode == 0) {
            float* outf = (float*)outp;
            #pragma unroll
            for (int v = 0; v < 8; ++v) {
                int M = hs * 8 + v;                  // D layout: M = hs*8 + v, N = lane%16
                outf[(m0 + M) * N + n] = acc[v] + bn;
            }
        } else if (mode == 1) {
            #pragma unroll
            for (int v = 0; v < 8; ++v) {
                int M = hs * 8 + v;
                Os[M * N + n] = acc[v] + bn;
            }
        } else {
            float* amps = (float*)outp;              // [BATCH][NBAND][TLEN]
            #pragma unroll
            for (int v = 0; v < 8; ++v) {
                int M = hs * 8 + v;
                int m = m0 + M;
                int b = m >> 11, t = m & (TLEN - 1);
                float x = acc[v] + bn;
                float s = 1.0f / (1.0f + __expf(-x));
                amps[(b * NBAND + n) * TLEN + t] =
                    2.0f * __powf(s, 2.302585092994046f) + 1e-18f;
            }
        }
    }

    if (mode == 1) {
        __syncthreads();
        if (tid < 16) {  // per-row LayerNorm statistics over N values
            float s = 0.0f, s2 = 0.0f;
            for (int j = 0; j < N; ++j) { float v = Os[tid * N + j]; s += v; s2 += v * v; }
            float m = s / (float)N;
            float var = s2 / (float)N - m * m;
            mrow[tid] = m;
            rrow[tid] = __frsqrt_rn(var + 1e-5f);
        }
        __syncthreads();
        bf16* outb = (bf16*)outp;
        for (int i = tid; i < 16 * N; i += 256) {
            int r = i / N, j = i % N;
            float y = (Os[r * N + j] - mrow[r]) * rrow[r] * gamma[j] + beta[j];
            y = (y >= 0.0f) ? y : 0.01f * y;
            outb[(m0 + r) * out_ld + j] = (bf16)y;
        }
    }
}

// ---------------------------------------------------------------------------
// GRU recurrence: 2048 strictly sequential steps. One workgroup, 384 threads.
// Whh^T kept f32-resident in LDS (196 KB of the 320 KB/WGP), layout
// WhhT[k*384 + j] so phase-1 reads are bank-conflict free (h[k] broadcasts).
// Emits the o0-layer input rows [h(128) | cp0 | cp1 | 0...] as bf16 [MTOT,160].
// ---------------------------------------------------------------------------
__global__ void gru_kernel(const float* __restrict__ Whh,  // [384][128]
                           const float* __restrict__ bhh,  // [384]
                           const float* __restrict__ xp,   // [MTOT][384] = x@Wih^T + bih
                           const float* __restrict__ cp0, const float* __restrict__ cp1,
                           bf16* __restrict__ act0) {      // [MTOT][160]
    extern __shared__ float sm[];
    float* WhhT = sm;                   // [128][384]
    float* hM   = WhhT + 128 * 384;     // [2][128]
    float* gh   = hM + 2 * 128;         // [2][384]
    const int tid = threadIdx.x;        // 0..383

    for (int i = tid; i < 384 * 128; i += 384) {
        int j = i / 128, k = i % 128;
        WhhT[k * 384 + j] = Whh[i];
    }
    for (int i = tid; i < 2 * 128; i += 384) hM[i] = 0.0f;
    __syncthreads();

    const float bj = bhh[tid];
    for (int t = 0; t < TLEN; ++t) {
        // phase 1: gh[b][j] = h[b] . Whh[j] + bhh[j]
        float a0 = bj, a1 = bj;
        #pragma unroll 4
        for (int k = 0; k < 128; ++k) {
            float w = WhhT[k * 384 + tid];
            a0 = fmaf(w, hM[k], a0);
            a1 = fmaf(w, hM[128 + k], a1);
        }
        gh[tid] = a0;
        gh[384 + tid] = a1;
        __syncthreads();
        // phase 2: gates + state update (256 threads: b = tid/128, j = tid%128)
        if (tid < 256) {
            int b = tid >> 7, j = tid & 127;
            const float* xr = xp + (b * TLEN + t) * 384;
            const float* gr = gh + b * 384;
            float r = 1.0f / (1.0f + __expf(-(xr[j]       + gr[j])));
            float z = 1.0f / (1.0f + __expf(-(xr[128 + j] + gr[128 + j])));
            float n = tanhf(xr[256 + j] + r * gr[256 + j]);
            float hn = (1.0f - z) * n + z * hM[b * 128 + j];
            hM[b * 128 + j] = hn;   // only this thread touches this slot in phase 2
            bf16* ar = act0 + (b * TLEN + t) * 160;
            ar[j] = (bf16)hn;
            if (j == 0) {
                ar[128] = (bf16)cp0[b * TLEN + t];
                ar[129] = (bf16)cp1[b * TLEN + t];
            }
        }
        __syncthreads();            // next phase-1 reads all of hM
    }
}

// ---------------------------------------------------------------------------
// Synthesis: out[b][s] = sum_c lerp(amps[b][c], s) * noise[c][s].
// Block of 256 samples stages its 10 amplitude frames (all bands, both
// batches) in LDS (40 KB); noise_bands (128 MB) is streamed coalesced exactly
// once — both batches accumulated in the same pass (HBM-roofline bound).
// ---------------------------------------------------------------------------
#define SYN_FRAMES 10
__global__ void synth_kernel(const float* __restrict__ amps,   // [2][NBAND][FRAME_LEN]
                             const float* __restrict__ noise,  // [NBAND][NOISE_LEN]
                             float* __restrict__ out) {        // [2][NOISE_LEN]
    __shared__ float tile[2 * NBAND * SYN_FRAMES];
    const int s0 = blockIdx.x * 256;
    const int tid = threadIdx.x;
    const int s = s0 + tid;

    float src0 = fmaxf((s0 + 0.5f) * (1.0f / SYNTH_WIN) - 0.5f, 0.0f);
    const int fbase = (int)floorf(src0);

    for (int i = tid; i < 2 * NBAND * SYN_FRAMES; i += 256) {
        int f = i % SYN_FRAMES;
        int cb = i / SYN_FRAMES;          // cb = c*2 + b
        int b = cb & 1, c = cb >> 1;
        int fr = min(fbase + f, FRAME_LEN - 1);
        tile[i] = amps[(b * NBAND + c) * FRAME_LEN + fr];
    }
    __syncthreads();

    float src = fmaxf((s + 0.5f) * (1.0f / SYNTH_WIN) - 0.5f, 0.0f);
    int i0 = (int)floorf(src);
    float w = src - (float)i0;
    int i1 = min(i0 + 1, FRAME_LEN - 1);
    int f0 = i0 - fbase;
    int f1 = min(i1 - fbase, SYN_FRAMES - 1);

    float acc0 = 0.0f, acc1 = 0.0f;
    for (int c = 0; c < NBAND; ++c) {
        float nb = noise[c * NOISE_LEN + s];
        const float* t0 = tile + (c * 2 + 0) * SYN_FRAMES;
        const float* t1 = tile + (c * 2 + 1) * SYN_FRAMES;
        float u0 = t0[f0] + w * (t0[f1] - t0[f0]);
        float u1 = t1[f0] + w * (t1[f1] - t1[f0]);
        acc0 = fmaf(u0, nb, acc0);
        acc1 = fmaf(u1, nb, acc1);
    }
    out[s] = acc0;
    out[NOISE_LEN + s] = acc1;
}

// ---------------------------------------------------------------------------
// Host orchestration
// ---------------------------------------------------------------------------
extern "C" void kernel_launch(void* const* d_in, const int* in_sizes, int n_in,
                              void* d_out, int out_size, void* d_ws, size_t ws_size,
                              hipStream_t stream) {
    (void)in_sizes; (void)n_in; (void)out_size; (void)ws_size;
    const float* cp0      = (const float*)d_in[0];
    const float* cp1      = (const float*)d_in[1];
    const float* in0_W    = (const float*)d_in[2];
    const float* in0_b    = (const float*)d_in[3];
    const float* in0_g    = (const float*)d_in[4];
    const float* in0_beta = (const float*)d_in[5];
    const float* in1_W    = (const float*)d_in[6];
    const float* in1_b    = (const float*)d_in[7];
    const float* in1_g    = (const float*)d_in[8];
    const float* in1_beta = (const float*)d_in[9];
    const float* gru_Wih  = (const float*)d_in[10];
    const float* gru_Whh  = (const float*)d_in[11];
    const float* gru_bih  = (const float*)d_in[12];
    const float* gru_bhh  = (const float*)d_in[13];
    const float* o0_W = (const float*)d_in[14]; const float* o0_b = (const float*)d_in[15];
    const float* o0_g = (const float*)d_in[16]; const float* o0_be = (const float*)d_in[17];
    const float* o1_W = (const float*)d_in[18]; const float* o1_b = (const float*)d_in[19];
    const float* o1_g = (const float*)d_in[20]; const float* o1_be = (const float*)d_in[21];
    const float* o2_W = (const float*)d_in[22]; const float* o2_b = (const float*)d_in[23];
    const float* o2_g = (const float*)d_in[24]; const float* o2_be = (const float*)d_in[25];
    const float* amp_W = (const float*)d_in[26];
    const float* amp_b = (const float*)d_in[27];
    const float* noise_bands = (const float*)d_in[28];

    // Workspace carve-up (256B aligned); total ~21.6 MB.
    char* ws = (char*)d_ws;
    size_t off = 0;
    auto carve = [&](size_t bytes) -> void* {
        void* p = ws + off;
        off = (off + bytes + 255) & ~(size_t)255;
        return p;
    };
    bf16*  hiddenB = (bf16*) carve((size_t)MTOT * 256 * 2);   // input-MLP concat, bf16
    bf16*  WihB    = (bf16*) carve((size_t)384 * 256 * 2);
    float* xp      = (float*)carve((size_t)MTOT * 384 * 4);   // GRU input projections
    bf16*  act0    = (bf16*) carve((size_t)MTOT * 160 * 2);   // [h|cp0|cp1|pad]
    bf16*  o0WB    = (bf16*) carve((size_t)128 * 160 * 2);
    bf16*  o1WB    = (bf16*) carve((size_t)128 * 128 * 2);
    bf16*  o2WB    = (bf16*) carve((size_t)128 * 128 * 2);
    bf16*  ampWB   = (bf16*) carve((size_t)512 * 128 * 2);
    bf16*  act1    = (bf16*) carve((size_t)MTOT * 128 * 2);
    bf16*  act2    = (bf16*) carve((size_t)MTOT * 128 * 2);
    bf16*  act3    = (bf16*) carve((size_t)MTOT * 128 * 2);
    float* amps    = (float*)carve((size_t)BATCH * NBAND * TLEN * 4);

    // --- weight conversion / padding -------------------------------------
    auto cvt = [&](const float* s, bf16* d, int rows, int sc, int dc) {
        int total = rows * dc;
        pad_convert_bf16_kernel<<<(total + 255) / 256, 256, 0, stream>>>(s, d, rows, sc, dc);
    };
    cvt(gru_Wih, WihB, 384, 256, 256);
    cvt(o0_W,    o0WB, 128, 130, 160);   // K padded 130 -> 160 (mult of 32)
    cvt(o1_W,    o1WB, 128, 128, 128);
    cvt(o2_W,    o2WB, 128, 128, 128);
    cvt(amp_W,   ampWB, 512, 128, 128);
    {
        int n = MTOT * 160;
        zero_bf16_kernel<<<(n + 255) / 256, 256, 0, stream>>>(act0, n);  // pad cols
    }

    // --- input MLPs -------------------------------------------------------
    input_mlp_kernel<<<MTOT, 128, 0, stream>>>(cp0, cp1,
        in0_W, in0_b, in0_g, in0_beta, in1_W, in1_b, in1_g, in1_beta, hiddenB);

    // --- xp = hidden @ Wih^T + bih  (WMMA, mode 0) ------------------------
    wmma_gemm_kernel<<<MTOT / 16, 256, 16 * 256 * 2, stream>>>(
        hiddenB, 256, WihB, gru_bih, nullptr, nullptr, (void*)xp, 256, 384, 384, 0);

    // --- GRU recurrence (sequential scan, LDS-resident Whh^T) -------------
    {
        size_t shm = (size_t)(128 * 384 + 2 * 128 + 2 * 384) * 4;  // 200704 B
        (void)hipFuncSetAttribute((const void*)gru_kernel,
                                  hipFuncAttributeMaxDynamicSharedMemorySize, (int)shm);
        gru_kernel<<<1, 384, shm, stream>>>(gru_Whh, gru_bhh, xp, cp0, cp1, act0);
    }

    // --- output MLP stack (WMMA + fused LN + LeakyReLU, mode 1) -----------
    wmma_gemm_kernel<<<MTOT / 16, 256, 16 * 160 * 2 + 16 * 128 * 4, stream>>>(
        act0, 160, o0WB, o0_b, o0_g, o0_be, (void*)act1, 160, 128, 128, 1);
    wmma_gemm_kernel<<<MTOT / 16, 256, 16 * 128 * 2 + 16 * 128 * 4, stream>>>(
        act1, 128, o1WB, o1_b, o1_g, o1_be, (void*)act2, 128, 128, 128, 1);
    wmma_gemm_kernel<<<MTOT / 16, 256, 16 * 128 * 2 + 16 * 128 * 4, stream>>>(
        act2, 128, o2WB, o2_b, o2_g, o2_be, (void*)act3, 128, 128, 128, 1);

    // --- amplitudes (WMMA + fused scale fn + transpose store, mode 2) -----
    wmma_gemm_kernel<<<MTOT / 16, 256, 16 * 128 * 2, stream>>>(
        act3, 128, ampWB, amp_b, nullptr, nullptr, (void*)amps, 128, 512, 512, 2);

    // --- synthesis (HBM-roofline: one pass over 128 MB noise_bands) -------
    synth_kernel<<<NOISE_LEN / 256, 256, 0, stream>>>(amps, noise_bands, (float*)d_out);
}
